// DeepSeekExpert_7894149890078
// MI455X (gfx1250) — 16-bit and 8-bit matrix operand order repaired and checked by reading, compiled, not run on hardware
//
#include <hip/hip_runtime.h>
#include <hip/hip_bf16.h>
#include <stdint.h>

#define S_DIM 8192
#define H_DIM 7168
#define F_DIM 2048
#define KB1   (H_DIM / 128)
#define NB1   (F_DIM / 128)

typedef __attribute__((ext_vector_type(16))) int    v16i;
typedef __attribute__((ext_vector_type(8)))  float  v8f;
typedef __attribute__((ext_vector_type(16))) __bf16 v16bf;

union FragFp8 { v16i v; uint32_t u[16]; };
union FragBf  { v16bf v; uint4 u[2]; };

#if __has_builtin(__builtin_amdgcn_global_load_async_to_lds_b128)
#define HAVE_ASYNC_LDS 1
#else
#define HAVE_ASYNC_LDS 0
#endif

typedef int async_v4i __attribute__((vector_size(16)));
typedef __attribute__((address_space(1))) async_v4i* async_gptr;
typedef __attribute__((address_space(3))) async_v4i* async_lptr;

__device__ __forceinline__ void async_copy16(void* lds_dst, const void* gsrc) {
#if HAVE_ASYNC_LDS
  __builtin_amdgcn_global_load_async_to_lds_b128(
      (async_gptr)(gsrc), (async_lptr)(lds_dst), 0, 0);
#else
  *(uint4*)lds_dst = *(const uint4*)gsrc;
#endif
}

template <int N>
__device__ __forceinline__ void wait_asynccnt() {
#if HAVE_ASYNC_LDS
#if __has_builtin(__builtin_amdgcn_s_wait_asynccnt)
  __builtin_amdgcn_s_wait_asynccnt(N);
#else
  asm volatile("s_wait_asynccnt %0" ::"n"(N) : "memory");
#endif
#endif
}

__device__ __forceinline__ float bf16_bits_to_f32(uint32_t b) {
  return __uint_as_float(b << 16);
}
__device__ __forceinline__ uint16_t f32_to_bf16_bits(float f) {
  uint32_t u = __float_as_uint(f);
  u += 0x7fffu + ((u >> 16) & 1u);
  return (uint16_t)(u >> 16);
}

#if __has_builtin(__builtin_amdgcn_cvt_pk_fp8_f32)
#define HAVE_HW_FP8 1
#endif

__device__ __forceinline__ uint32_t f32_to_e4m3_bits(float x) {
  uint32_t s = (__float_as_uint(x) >> 31) << 7;
  float ax = fabsf(x);
  if (!(ax > 0.0f)) return s;
  ax = fminf(ax, 448.0f);
  int e;
  float m = frexpf(ax, &e);
  int E = e - 1 + 7;
  uint32_t bits;
  if (E >= 1) {
    float m2 = m * 2.0f;
    int frac = (int)rintf((m2 - 1.0f) * 8.0f);
    if (frac == 8) { frac = 0; ++E; }
    if (E > 15) { E = 15; frac = 7; }
    bits = ((uint32_t)E << 3) | (uint32_t)frac;
  } else {
    int frac = (int)rintf(ax * 512.0f);
    if (frac > 7) frac = 7;
    bits = (uint32_t)frac;
  }
  return s | bits;
}

__device__ __forceinline__ uint32_t pack4_e4m3(float f0, float f1, float f2, float f3) {
#ifdef HAVE_HW_FP8
  int p = __builtin_amdgcn_cvt_pk_fp8_f32(f0, f1, 0, false);
  p = __builtin_amdgcn_cvt_pk_fp8_f32(f2, f3, p, true);
  return (uint32_t)p;
#else
  return f32_to_e4m3_bits(f0) | (f32_to_e4m3_bits(f1) << 8) |
         (f32_to_e4m3_bits(f2) << 16) | (f32_to_e4m3_bits(f3) << 24);
#endif
}

__global__ __launch_bounds__(256) void q_token_kernel(
    const uint16_t* __restrict__ x, uint8_t* __restrict__ xq, float* __restrict__ xs) {
  const int row  = blockIdx.x;
  const int wave = threadIdx.x >> 5;
  const int lane = threadIdx.x & 31;
  const uint16_t* xrow = x + (size_t)row * H_DIM;
  uint8_t* qrow = xq + (size_t)row * H_DIM;

  for (int kb = wave; kb < KB1; kb += 8) {
    const int base = kb * 128 + lane * 4;
    uint2 raw = *(const uint2*)(xrow + base);
    float f0 = bf16_bits_to_f32(raw.x & 0xffffu);
    float f1 = bf16_bits_to_f32(raw.x >> 16);
    float f2 = bf16_bits_to_f32(raw.y & 0xffffu);
    float f3 = bf16_bits_to_f32(raw.y >> 16);
    float amax = fmaxf(fmaxf(fabsf(f0), fabsf(f1)), fmaxf(fabsf(f2), fabsf(f3)));
    #pragma unroll
    for (int off = 16; off > 0; off >>= 1)
      amax = fmaxf(amax, __shfl_xor(amax, off, 32));
    amax = fmaxf(amax, 1e-4f);
    float scale = amax * (1.0f / 448.0f);
    float inv   = 448.0f / amax;
    *(uint32_t*)(qrow + base) = pack4_e4m3(f0 * inv, f1 * inv, f2 * inv, f3 * inv);
    if (lane == 0) xs[(size_t)row * KB1 + kb] = scale;
  }
}

__global__ __launch_bounds__(256) void q_block_kernel(
    const uint16_t* __restrict__ w, uint8_t* __restrict__ wq, float* __restrict__ wsc) {
  const int nb  = blockIdx.x;
  const int mb  = blockIdx.y;
  const int tid = threadIdx.x;
  __shared__ float red[9];

  uint4 vals[8];
  float amax = 0.0f;
  #pragma unroll
  for (int it = 0; it < 8; ++it) {
    int L = it * 2048 + tid * 8;
    int r = L >> 7, c = L & 127;
    vals[it] = *(const uint4*)(w + (size_t)(mb * 128 + r) * H_DIM + nb * 128 + c);
    uint32_t d[4] = {vals[it].x, vals[it].y, vals[it].z, vals[it].w};
    #pragma unroll
    for (int j = 0; j < 4; ++j) {
      amax = fmaxf(amax, fabsf(bf16_bits_to_f32(d[j] & 0xffffu)));
      amax = fmaxf(amax, fabsf(bf16_bits_to_f32(d[j] >> 16)));
    }
  }
  #pragma unroll
  for (int off = 16; off > 0; off >>= 1)
    amax = fmaxf(amax, __shfl_xor(amax, off, 32));
  if ((tid & 31) == 0) red[tid >> 5] = amax;
  __syncthreads();
  if (tid == 0) {
    float m = red[0];
    #pragma unroll
    for (int i = 1; i < 8; ++i) m = fmaxf(m, red[i]);
    red[8] = fmaxf(m, 1e-4f);
  }
  __syncthreads();
  float amaxAll = red[8];
  float inv = 448.0f / amaxAll;
  if (tid == 0) wsc[mb * KB1 + nb] = amaxAll * (1.0f / 448.0f);

  #pragma unroll
  for (int it = 0; it < 8; ++it) {
    int L = it * 2048 + tid * 8;
    int r = L >> 7, c = L & 127;
    uint32_t d[4] = {vals[it].x, vals[it].y, vals[it].z, vals[it].w};
    float f[8];
    #pragma unroll
    for (int j = 0; j < 4; ++j) {
      f[2 * j]     = bf16_bits_to_f32(d[j] & 0xffffu) * inv;
      f[2 * j + 1] = bf16_bits_to_f32(d[j] >> 16) * inv;
    }
    uint2 q;
    q.x = pack4_e4m3(f[0], f[1], f[2], f[3]);
    q.y = pack4_e4m3(f[4], f[5], f[6], f[7]);
    *(uint2*)(wq + (size_t)(mb * 128 + r) * H_DIM + nb * 128 + c) = q;
  }
}

__global__ __launch_bounds__(256) void gemm1_fp8_kernel(
    const uint8_t* __restrict__ xq, const float* __restrict__ xs,
    const uint8_t* __restrict__ wq, const float* __restrict__ wsc,
    uint16_t* __restrict__ h) {
  __shared__ uint8_t ldsA[2][128 * 128];
  __shared__ uint8_t ldsB[2][128 * 128];
  __shared__ float lds_xs[2][128];

  const int n0 = blockIdx.x * 128;
  const int m0 = blockIdx.y * 128;
  const int tid   = threadIdx.x;
  const int wave  = tid >> 5;
  const int lane  = tid & 31;
  const int waveM = wave >> 1;
  const int waveN = wave & 1;
  const int laneM  = lane & 15;
  const int laneHi = lane >> 4;
  const int nb56 = (n0 >> 7) * KB1;

  v8f acc[2][4] = {};

  auto issue_slab = [&](int kb, int buf) {
    #pragma unroll
    for (int it = 0; it < 4; ++it) {
      int L = it * 4096 + tid * 16;
      int r = L >> 7, c = L & 127;
      async_copy16(&ldsA[buf][r * 128 + c],
                   xq + (size_t)(m0 + r) * H_DIM + kb * 128 + c);
      async_copy16(&ldsB[buf][r * 128 + c],
                   wq + (size_t)(n0 + r) * H_DIM + kb * 128 + c);
    }
    if (tid < 128) lds_xs[buf][tid] = xs[(size_t)(m0 + tid) * KB1 + kb];
  };

  issue_slab(0, 0);

  for (int kb = 0; kb < KB1; ++kb) {
    const int buf = kb & 1;
    if (kb + 1 < KB1) {
      issue_slab(kb + 1, buf ^ 1);
      wait_asynccnt<8>();
    } else {
      wait_asynccnt<0>();
    }
    float wskb = wsc[nb56 + kb];
    __syncthreads();

    float sA[2][8];
    #pragma unroll
    for (int mi = 0; mi < 2; ++mi)
      #pragma unroll
      for (int e = 0; e < 8; ++e)
        sA[mi][e] = lds_xs[buf][waveM * 32 + mi * 16 + laneHi * 8 + e] * wskb;

    FragFp8 afrag[2];
    #pragma unroll
    for (int mi = 0; mi < 2; ++mi) {
      const uint8_t* arow = &ldsA[buf][(waveM * 32 + mi * 16 + laneM) * 128];
      const int kofs = laneHi * 8;
      #pragma unroll
      for (int c = 0; c < 8; ++c) {
        uint2 d = *(const uint2*)(arow + 32 * (c >> 1) + 8 * (c & 1) + 2 * kofs);
        afrag[mi].u[2 * c] = d.x;
        afrag[mi].u[2 * c + 1] = d.y;
      }
    }

    #pragma unroll
    for (int ni = 0; ni < 4; ++ni) {
      FragFp8 bfrag;
      const uint8_t* brow = &ldsB[buf][(waveN * 64 + ni * 16 + laneM) * 128];
      const int kb16 = laneHi * 16;
      #pragma unroll
      for (int g = 0; g < 4; ++g) {
        uint4 d = *(const uint4*)(brow + 32 * g + kb16);
        bfrag.u[4 * g] = d.x; bfrag.u[4 * g + 1] = d.y;
        bfrag.u[4 * g + 2] = d.z; bfrag.u[4 * g + 3] = d.w;
      }
      #pragma unroll
      for (int mi = 0; mi < 2; ++mi) {
        v8f z = {};
        v8f part = __builtin_amdgcn_wmma_f32_16x16x128_fp8_fp8(
            afrag[mi].v, bfrag.v, (short)0, z, false, false);
        #pragma unroll
        for (int e = 0; e < 8; ++e)
          acc[mi][ni][e] += part[e] * sA[mi][e];
      }
    }
    __syncthreads();
  }

  #pragma unroll
  for (int mi = 0; mi < 2; ++mi)
    #pragma unroll
    for (int ni = 0; ni < 4; ++ni)
      #pragma unroll
      for (int e = 0; e < 8; ++e) {
        int row = m0 + waveM * 32 + mi * 16 + laneHi * 8 + e;
        int col = n0 + waveN * 64 + ni * 16 + laneM;
        float o = acc[mi][ni][e];
        float v = o * o * (1.0f / (1.0f + __expf(-o)));
        h[(size_t)row * F_DIM + col] = f32_to_bf16_bits(v);
      }
}

__global__ __launch_bounds__(256) void gemm2_bf16_kernel(
    const uint16_t* __restrict__ hmat, const uint16_t* __restrict__ w2,
    uint16_t* __restrict__ out) {
  __shared__ uint16_t ldsA[2][128 * 64];
  __shared__ uint16_t ldsB[2][128 * 64];

  const int n0 = blockIdx.x * 128;
  const int m0 = blockIdx.y * 128;
  const int tid   = threadIdx.x;
  const int wave  = tid >> 5;
  const int lane  = tid & 31;
  const int waveM = wave >> 1;
  const int waveN = wave & 1;
  const int laneM  = lane & 15;
  const int laneHi = lane >> 4;

  v8f acc[2][4] = {};

  auto issue_slab = [&](int k0, int buf) {
    #pragma unroll
    for (int it = 0; it < 4; ++it) {
      int L = it * 4096 + tid * 16;
      int r = L >> 7, c = L & 127;
      async_copy16((uint8_t*)ldsA[buf] + r * 128 + c,
                   (const uint8_t*)(hmat + (size_t)(m0 + r) * F_DIM + k0) + c);
      async_copy16((uint8_t*)ldsB[buf] + r * 128 + c,
                   (const uint8_t*)(w2 + (size_t)(n0 + r) * F_DIM + k0) + c);
    }
  };

  issue_slab(0, 0);

  for (int k0 = 0; k0 < F_DIM; k0 += 64) {
    const int buf = (k0 >> 6) & 1;
    if (k0 + 64 < F_DIM) {
      issue_slab(k0 + 64, buf ^ 1);
      wait_asynccnt<8>();
    } else {
      wait_asynccnt<0>();
    }
    __syncthreads();

    #pragma unroll
    for (int ks = 0; ks < 64; ks += 32) {
      FragBf afrag[2];
      #pragma unroll
      for (int mi = 0; mi < 2; ++mi) {
        const uint8_t* arow =
            (const uint8_t*)&ldsA[buf][(waveM * 32 + mi * 16 + laneM) * 64 + ks];
        afrag[mi].u[0] = *(const uint4*)(arow + laneHi * 16);
        afrag[mi].u[1] = *(const uint4*)(arow + 32 + laneHi * 16);
      }
      #pragma unroll
      for (int ni = 0; ni < 4; ++ni) {
        FragBf bfrag;
        const uint8_t* brow =
            (const uint8_t*)&ldsB[buf][(waveN * 64 + ni * 16 + laneM) * 64 + ks];
        bfrag.u[0] = *(const uint4*)(brow + laneHi * 16);
        bfrag.u[1] = *(const uint4*)(brow + 32 + laneHi * 16);
        #pragma unroll
        for (int mi = 0; mi < 2; ++mi) {
          acc[mi][ni] = __builtin_amdgcn_wmma_f32_16x16x32_bf16(
              false, afrag[mi].v, false, bfrag.v, (short)0, acc[mi][ni], false, false);
        }
      }
    }
    __syncthreads();
  }

  #pragma unroll
  for (int mi = 0; mi < 2; ++mi)
    #pragma unroll
    for (int ni = 0; ni < 4; ++ni)
      #pragma unroll
      for (int e = 0; e < 8; ++e) {
        int row = m0 + waveM * 32 + mi * 16 + laneHi * 8 + e;
        int col = n0 + waveN * 64 + ni * 16 + laneM;
        out[(size_t)row * H_DIM + col] = f32_to_bf16_bits(acc[mi][ni][e]);
      }
}

extern "C" void kernel_launch(void* const* d_in, const int* in_sizes, int n_in,
                              void* d_out, int out_size, void* d_ws, size_t ws_size,
                              hipStream_t stream) {
  (void)in_sizes; (void)n_in; (void)out_size; (void)ws_size;
  const uint16_t* x  = (const uint16_t*)d_in[0];
  const uint16_t* w1 = (const uint16_t*)d_in[1];
  const uint16_t* w2 = (const uint16_t*)d_in[2];
  uint16_t* out = (uint16_t*)d_out;

  uint8_t* ws = (uint8_t*)d_ws;
  size_t off = 0;
  uint8_t*  xq  = ws + off;            off += (size_t)S_DIM * H_DIM;
  float*    xs  = (float*)(ws + off);  off += (size_t)S_DIM * KB1 * 4;
  uint8_t*  w1q = ws + off;            off += (size_t)F_DIM * H_DIM;
  float*    w1s = (float*)(ws + off);  off += (size_t)NB1 * KB1 * 4;
  uint16_t* h   = (uint16_t*)(ws + off);

  q_token_kernel<<<S_DIM, 256, 0, stream>>>(x, xq, xs);

  dim3 gq(KB1, NB1);
  q_block_kernel<<<gq, 256, 0, stream>>>(w1, w1q, w1s);

  dim3 g1(F_DIM / 128, S_DIM / 128);
  gemm1_fp8_kernel<<<g1, 256, 0, stream>>>(xq, xs, w1q, w1s, h);

  dim3 g2(H_DIM / 128, S_DIM / 128);
  gemm2_bf16_kernel<<<g2, 256, 0, stream>>>(h, w2, out);
}
